// CausalSelfAttention_19645180412429
// MI455X (gfx1250) — matrix operand order repaired and checked by reading, not compiled, not run on hardware
//
#include <hip/hip_runtime.h>
#include <math.h>

// ---------------------------------------------------------------------------
// CausalSelfAttention for MI455X (gfx1250): bf16 WMMA flash attention.
//   B=2, T=4096, C=512, H=8, hd=64.
// Pipeline:
//   0) one-pass fp32->bf16 conversion of x and the 4 weight matrices
//   1) fused QKV GEMM, 32x32 tile per wave (pure b128 loads -> wmma)
//   2) flash attention, one wave per (b,h,16-query tile)
//   3) output projection GEMM, 32x32 tile per wave, fp32 out + bias
// ---------------------------------------------------------------------------

typedef __bf16 bf16;
typedef __attribute__((ext_vector_type(16))) __bf16 v16bf;
typedef __attribute__((ext_vector_type(8)))  __bf16 v8bf;
typedef __attribute__((ext_vector_type(8)))  float  v8f;

#define CDIM 512
#define HN   8
#define HD   64
#define BN   2
#define TN   4096
#define BT   (BN * TN)      // 8192 rows

// D = A(16x32) * B(32x16) + C, bf16 in, f32 accumulate
__device__ __forceinline__ v8f wmma_bf16(v16bf a, v16bf b, v8f c) {
  return __builtin_amdgcn_wmma_f32_16x16x32_bf16(
      /*neg_a=*/false, a, /*neg_b=*/false, b,
      /*c_mod=*/(short)0, c, /*reuse_a=*/false, /*reuse_b=*/false);
}

// ---- fragment loaders (bf16 only; contiguous k order, same in A and B) -------
// A-matrix 16x32:
//   lanes 0-15 : M=lane,    elems 0..15 = K k0..k0+15   (contiguous)
//   lanes16-31 : M=lane-16, elems 0..15 = K k0+16..+31  (contiguous)
__device__ __forceinline__ v16bf load_a_bf16(const bf16* __restrict__ base,
                                             int ld, int m0, int k0, int lane) {
  int m  = m0 + (lane & 15);
  int kk = k0 + ((lane >> 4) << 4);
  const bf16* p = base + (size_t)m * ld + kk;
  union { v16bf v; v8bf h[2]; } u;
  u.h[0] = *(const v8bf*)(p);
  u.h[1] = *(const v8bf*)(p + 8);
  return u.v;
}

// B-matrix 32x16:
//   lanes 0-15 : N=lane,    elems 0..15 = K k0..k0+15   (contiguous)
//   lanes16-31 : N=lane-16, elems 0..15 = K k0+16..+31  (contiguous)
// For y = x @ W^T, B[k,n] = W[n,k] -> contiguous along W's row.
__device__ __forceinline__ v16bf load_b_bf16(const bf16* __restrict__ W,
                                             int ld, int n0, int k0, int lane) {
  int n  = n0 + (lane & 15);
  int kk = k0 + ((lane >> 4) << 4);
  const bf16* p = W + (size_t)n * ld + kk;
  union { v16bf v; v8bf h[2]; } u;
  u.h[0] = *(const v8bf*)(p);
  u.h[1] = *(const v8bf*)(p + 8);
  return u.v;
}

// ---------------------------------------------------------------------------
// Kernel 0: fp32 -> bf16 one-pass conversion (8 elems / thread, packed store)
// ---------------------------------------------------------------------------
__global__ __launch_bounds__(256) void cvt_f32_to_bf16_kernel(
    const float* __restrict__ src, bf16* __restrict__ dst, int n8) {
  int i = blockIdx.x * blockDim.x + threadIdx.x;
  if (i < n8) {
    const float* p = src + (size_t)i * 8;
    v8bf o;
#pragma unroll
    for (int r = 0; r < 8; ++r) o[r] = (bf16)p[r];
    *(v8bf*)(dst + (size_t)i * 8) = o;
  }
}

// ---------------------------------------------------------------------------
// Kernel 1: fused QKV projection.  [8192 x 512] @ [512 x 1536], 32x32/wave.
// Q,K stored bf16 [B,T,C].  V stored transposed bf16 [B,H,HD,T].
// ---------------------------------------------------------------------------
__global__ __launch_bounds__(256) void qkv_proj_kernel(
    const bf16* __restrict__ xb,
    const bf16* __restrict__ Wqb, const bf16* __restrict__ Wkb,
    const bf16* __restrict__ Wvb,
    const float* __restrict__ bq, const float* __restrict__ bk,
    const float* __restrict__ bv,
    bf16* __restrict__ Qb, bf16* __restrict__ Kb, bf16* __restrict__ Vt) {
  const int lane = threadIdx.x & 31;
  const int wave = blockIdx.x * (blockDim.x >> 5) + (threadIdx.x >> 5);
  const int nTiles = (3 * CDIM) / 32;          // 48
  const int mT = wave / nTiles;
  const int nT = wave % nTiles;
  const int m0 = mT * 32;
  const int ng0 = nT * 32;                     // 0..1535, never crosses 512-bounds
  const int which = ng0 / CDIM;                // 0=Q 1=K 2=V
  const int n0 = ng0 % CDIM;

  const bf16*  W    = (which == 0) ? Wqb : (which == 1) ? Wkb : Wvb;
  const float* bias = (which == 0) ? bq  : (which == 1) ? bk  : bv;

  v8f a00 = {}, a01 = {}, a10 = {}, a11 = {};
#pragma unroll 4
  for (int k0 = 0; k0 < CDIM; k0 += 32) {
    v16bf af0 = load_a_bf16(xb, CDIM, m0,      k0, lane);
    v16bf af1 = load_a_bf16(xb, CDIM, m0 + 16, k0, lane);
    v16bf bf0 = load_b_bf16(W,  CDIM, n0,      k0, lane);
    v16bf bf1 = load_b_bf16(W,  CDIM, n0 + 16, k0, lane);
    a00 = wmma_bf16(af0, bf0, a00);
    a01 = wmma_bf16(af0, bf1, a01);
    a10 = wmma_bf16(af1, bf0, a10);
    a11 = wmma_bf16(af1, bf1, a11);
  }

  // C/D layout: lane = N; VGPR r -> row m0+r (lanes 0-15) / m0+8+r (lanes 16-31)
  const int hiOff = (lane >> 4) << 3;
  v8f* accs[4] = {&a00, &a01, &a10, &a11};
#pragma unroll
  for (int t = 0; t < 4; ++t) {
    const int i = t >> 1, j = t & 1;           // sub-tile (m,n) indices
    const int n = n0 + j * 16 + (lane & 15);
    const int rbase = m0 + i * 16 + hiOff;
    const float bn = bias[n];
    v8f acc = *accs[t];
    if (which < 2) {
      bf16* dst = (which == 0) ? Qb : Kb;
#pragma unroll
      for (int r = 0; r < 8; ++r)
        dst[(size_t)(rbase + r) * CDIM + n] = (bf16)(acc[r] + bn);
    } else {
      // V^T[b,h,d,t]: lane holds fixed (h,d), 8 consecutive tokens -> packed
      const int h = n / HD, d = n % HD;
      const int bI = rbase / TN, tt = rbase % TN;
      bf16* p = Vt + (((size_t)(bI * HN + h) * HD + d) * TN + tt);
      v8bf pk;
#pragma unroll
      for (int r = 0; r < 8; ++r) pk[r] = (bf16)(acc[r] + bn);
      *(v8bf*)p = pk;
    }
  }
}

// ---------------------------------------------------------------------------
// Kernel 2: flash attention.  One wave per (b, h, 16-query tile).
// S^T = K * Q^T (two 16x16 f32 tiles / 32-key block), online softmax,
// O^T += V^T * P^T (four 16x16 f32 tiles over head-dim).
// ---------------------------------------------------------------------------
__global__ __launch_bounds__(256) void flash_attn_kernel(
    const bf16* __restrict__ Qb, const bf16* __restrict__ Kb,
    const bf16* __restrict__ Vt, bf16* __restrict__ Yb) {
  const int lane = threadIdx.x & 31;
  const int wave = blockIdx.x * (blockDim.x >> 5) + (threadIdx.x >> 5);
  const int qtiles = TN / 16;                   // 256
  const int bh = wave / qtiles;
  const int qt = wave % qtiles;
  const int b = bh / HN, h = bh % HN;
  const int q0 = qt * 16;

  const bf16* Qh = Qb + (size_t)b * TN * CDIM + h * HD;   // row stride CDIM
  const bf16* Kh = Kb + (size_t)b * TN * CDIM + h * HD;   // row stride CDIM
  const bf16* Vh = Vt + (size_t)(b * HN + h) * HD * TN;   // [d][t], stride TN

  const bool hi = lane >= 16;
  const int  q  = q0 + (lane & 15);

  // Q^T B-fragments (register-resident across the key loop)
  v16bf qf[2];
  {
    const bf16* qp = Qh + (size_t)q * CDIM + ((lane >> 4) << 4);
    union { v16bf v; v8bf h2[2]; } u0, u1;
    u0.h2[0] = *(const v8bf*)(qp);
    u0.h2[1] = *(const v8bf*)(qp + 8);
    u1.h2[0] = *(const v8bf*)(qp + 32);
    u1.h2[1] = *(const v8bf*)(qp + 40);
    qf[0] = u0.v;                               // d 0..31
    qf[1] = u1.v;                               // d 32..63
  }

  v8f o0 = {}, o1 = {}, o2 = {}, o3 = {};
  float m_run = -3.0e38f;
  float l_run = 0.0f;
  const float scale = 0.125f;                   // 1/sqrt(64)

  const int kend = q0 + 16;                     // causal: keys <= q0+15
  for (int kb = 0; kb < kend; kb += 32) {
    if (kb + 32 < kend)                         // prefetch next K block
      __builtin_prefetch(Kh + (size_t)(kb + 32 + (lane & 15)) * CDIM, 0, 1);

    // --- S^T tiles: keys kb..kb+15 (s0) and kb+16..kb+31 (s1) vs 16 queries
    v8f s0 = {}, s1 = {};
#pragma unroll
    for (int k0 = 0; k0 < HD; k0 += 32) {
      v16bf ka0 = load_a_bf16(Kh, CDIM, kb,      k0, lane);
      v16bf ka1 = load_a_bf16(Kh, CDIM, kb + 16, k0, lane);
      v16bf qb  = qf[k0 >> 5];
      s0 = wmma_bf16(ka0, qb, s0);
      s1 = wmma_bf16(ka1, qb, s1);
    }

    // --- scale + causal mask (C layout: lane = query, VGPR r = key row)
    const int kb0 = kb + (hi ? 8 : 0);
#pragma unroll
    for (int r = 0; r < 8; ++r) {
      float v0 = s0[r] * scale;
      float v1 = s1[r] * scale;
      if (kb0 + r      > q) v0 = -3.0e38f;
      if (kb0 + 16 + r > q) v1 = -3.0e38f;
      s0[r] = v0; s1[r] = v1;
    }

    // --- online softmax stats (per query = per lane; combine halves)
    float bm = -3.0e38f;
#pragma unroll
    for (int r = 0; r < 8; ++r) bm = fmaxf(bm, fmaxf(s0[r], s1[r]));
    bm = fmaxf(bm, __shfl_xor(bm, 16, 32));
    const float m_new = fmaxf(m_run, bm);
    const float alpha = __expf(m_run - m_new);

    v8f p0, p1;
    float lsum = 0.0f;
#pragma unroll
    for (int r = 0; r < 8; ++r) {
      float e0 = __expf(s0[r] - m_new);
      float e1 = __expf(s1[r] - m_new);
      p0[r] = e0; p1[r] = e1;
      lsum += e0 + e1;
    }
    lsum += __shfl_xor(lsum, 16, 32);
    l_run = l_run * alpha + lsum;
    m_run = m_new;

    // --- rescale O accumulators by alpha (per-lane = per-query)
#pragma unroll
    for (int r = 0; r < 8; ++r) {
      o0[r] *= alpha; o1[r] *= alpha; o2[r] *= alpha; o3[r] *= alpha;
    }

    // --- S^T C-layout -> P^T B-layout: one half-swap shuffle per register.
    // lanes 0-15 want keys kb+0..15 : [own s0 | s0 from lane^16]
    // lanes16-31 want keys kb+16..31: [s1 from lane^16 | own s1]
    union { v16bf v; bf16 e[16]; } pb;
#pragma unroll
    for (int r = 0; r < 8; ++r) {
      float oth0 = __shfl_xor(p0[r], 16, 32);
      float oth1 = __shfl_xor(p1[r], 16, 32);
      pb.e[r]     = (bf16)(hi ? oth1 : p0[r]);
      pb.e[r + 8] = (bf16)(hi ? p1[r] : oth0);
    }

    // --- O^T += V^T(16d x 32k) * P^T(32k x 16q), four head-dim tiles
    v16bf va0 = load_a_bf16(Vh, TN,  0, kb, lane);
    v16bf va1 = load_a_bf16(Vh, TN, 16, kb, lane);
    v16bf va2 = load_a_bf16(Vh, TN, 32, kb, lane);
    v16bf va3 = load_a_bf16(Vh, TN, 48, kb, lane);
    o0 = wmma_bf16(va0, pb.v, o0);
    o1 = wmma_bf16(va1, pb.v, o1);
    o2 = wmma_bf16(va2, pb.v, o2);
    o3 = wmma_bf16(va3, pb.v, o3);
  }

  // --- finalize: O^T C-layout -> lane holds 8 consecutive d values per tile
  const float inv = 1.0f / l_run;
  bf16* yp = Yb + ((size_t)(b * TN + q)) * CDIM + h * HD + ((lane >> 4) << 3);
  v8f* ot[4] = {&o0, &o1, &o2, &o3};
#pragma unroll
  for (int dt = 0; dt < 4; ++dt) {
    v8bf pk;
#pragma unroll
    for (int r = 0; r < 8; ++r) pk[r] = (bf16)((*ot[dt])[r] * inv);
    *(v8bf*)(yp + dt * 16) = pk;
  }
}

// ---------------------------------------------------------------------------
// Kernel 3: output projection.  [8192x512](bf16) @ Wp^T + bp -> fp32, 32x32/wave
// ---------------------------------------------------------------------------
__global__ __launch_bounds__(256) void out_proj_kernel(
    const bf16* __restrict__ Yb, const bf16* __restrict__ Wpb,
    const float* __restrict__ bp, float* __restrict__ out) {
  const int lane = threadIdx.x & 31;
  const int wave = blockIdx.x * (blockDim.x >> 5) + (threadIdx.x >> 5);
  const int nTiles = CDIM / 32;                // 16
  const int mT = wave / nTiles;
  const int nT = wave % nTiles;
  const int m0 = mT * 32;
  const int n0 = nT * 32;

  v8f a00 = {}, a01 = {}, a10 = {}, a11 = {};
#pragma unroll 4
  for (int k0 = 0; k0 < CDIM; k0 += 32) {
    v16bf af0 = load_a_bf16(Yb,  CDIM, m0,      k0, lane);
    v16bf af1 = load_a_bf16(Yb,  CDIM, m0 + 16, k0, lane);
    v16bf bf0 = load_b_bf16(Wpb, CDIM, n0,      k0, lane);
    v16bf bf1 = load_b_bf16(Wpb, CDIM, n0 + 16, k0, lane);
    a00 = wmma_bf16(af0, bf0, a00);
    a01 = wmma_bf16(af0, bf1, a01);
    a10 = wmma_bf16(af1, bf0, a10);
    a11 = wmma_bf16(af1, bf1, a11);
  }

  const int hiOff = (lane >> 4) << 3;
  v8f* accs[4] = {&a00, &a01, &a10, &a11};
#pragma unroll
  for (int t = 0; t < 4; ++t) {
    const int i = t >> 1, j = t & 1;
    const int n = n0 + j * 16 + (lane & 15);
    const int rbase = m0 + i * 16 + hiOff;
    const float bn = bp[n];
    v8f acc = *accs[t];
#pragma unroll
    for (int r = 0; r < 8; ++r)
      out[(size_t)(rbase + r) * CDIM + n] = acc[r] + bn;
  }
}

// ---------------------------------------------------------------------------
extern "C" void kernel_launch(void* const* d_in, const int* in_sizes, int n_in,
                              void* d_out, int out_size, void* d_ws, size_t ws_size,
                              hipStream_t stream) {
  (void)in_sizes; (void)n_in; (void)out_size; (void)ws_size;

  const float* x  = (const float*)d_in[0];
  // d_in[1] = weight, d_in[2] = state : unused by the reference math
  const float* Wq = (const float*)d_in[3];
  const float* bq = (const float*)d_in[4];
  const float* Wk = (const float*)d_in[5];
  const float* bk = (const float*)d_in[6];
  const float* Wv = (const float*)d_in[7];
  const float* bv = (const float*)d_in[8];
  const float* Wp = (const float*)d_in[9];
  const float* bp = (const float*)d_in[10];
  float* out = (float*)d_out;

  const size_t planeElems = (size_t)BT * CDIM;   // 4,194,304 bf16 = 8 MB
  const size_t wElems     = (size_t)CDIM * CDIM; //   262,144 bf16 = 512 KB
  bf16* Qb  = (bf16*)d_ws;
  bf16* Kb  = Qb  + planeElems;
  bf16* Vt  = Kb  + planeElems;
  bf16* Yb  = Vt  + planeElems;
  bf16* xbb = Yb  + planeElems;
  bf16* Wqb = xbb + planeElems;
  bf16* Wkb = Wqb + wElems;
  bf16* Wvb = Wkb + wElems;
  bf16* Wpb = Wvb + wElems;

  // 0) one-pass fp32 -> bf16 conversions (all L2-resident afterwards)
  cvt_f32_to_bf16_kernel<<<(int)(planeElems / 8 / 256), 256, 0, stream>>>(x,  xbb, (int)(planeElems / 8));
  cvt_f32_to_bf16_kernel<<<(int)(wElems / 8 / 256),     256, 0, stream>>>(Wq, Wqb, (int)(wElems / 8));
  cvt_f32_to_bf16_kernel<<<(int)(wElems / 8 / 256),     256, 0, stream>>>(Wk, Wkb, (int)(wElems / 8));
  cvt_f32_to_bf16_kernel<<<(int)(wElems / 8 / 256),     256, 0, stream>>>(Wv, Wvb, (int)(wElems / 8));
  cvt_f32_to_bf16_kernel<<<(int)(wElems / 8 / 256),     256, 0, stream>>>(Wp, Wpb, (int)(wElems / 8));

  // 1) QKV projection: (8192/32) * (1536/32) = 12288 waves, 8 waves/block
  qkv_proj_kernel<<<1536, 256, 0, stream>>>(xbb, Wqb, Wkb, Wvb, bq, bk, bv,
                                            Qb, Kb, Vt);

  // 2) Flash attention: B*H*(T/16) = 4096 waves
  flash_attn_kernel<<<512, 256, 0, stream>>>(Qb, Kb, Vt, Yb);

  // 3) Output projection: (8192/32) * (512/32) = 4096 waves
  out_proj_kernel<<<512, 256, 0, stream>>>(Yb, Wpb, bp, out);
}
